// ExpertsFeedForward_5454608466018
// MI455X (gfx1250) — matrix operand order repaired and checked by reading, not compiled, not run on hardware
//
#include <hip/hip_runtime.h>

// ---------------- static problem config ----------------
#define T_TOK 16384   // G*S
#define DIM   1024    // D
#define HID   4096    // H
#define NEXP  8       // E
#define CAP   2048    // capacity per expert

typedef __attribute__((ext_vector_type(16))) __bf16         v16bf;
typedef __attribute__((ext_vector_type(8)))  float          v8f;
typedef __attribute__((ext_vector_type(8)))  unsigned short ushort8;
typedef int v4i_g __attribute__((__vector_size__(4 * sizeof(int))));

#define AS1 __attribute__((address_space(1)))
#define AS3 __attribute__((address_space(3)))

#if __has_builtin(__builtin_amdgcn_global_load_async_to_lds_b128)
#define HAVE_ASYNC_LDS 1
#else
#define HAVE_ASYNC_LDS 0
#endif

__device__ __forceinline__ void async_b128(const void* g, void* l) {
#if HAVE_ASYNC_LDS
  __builtin_amdgcn_global_load_async_to_lds_b128(
      (AS1 v4i_g*)(AS1 void*)(void*)g, (AS3 v4i_g*)(AS3 void*)l, 0, 0);
#else
  (void)g; (void)l;
#endif
}
__device__ __forceinline__ void wait_async0() {
#if HAVE_ASYNC_LDS
#if __has_builtin(__builtin_amdgcn_s_wait_asynccnt)
  __builtin_amdgcn_s_wait_asynccnt(0);
#else
  asm volatile("s_wait_asynccnt 0x0" ::: "memory");
#endif
#endif
}
__device__ __forceinline__ void sched_fence() {
#if __has_builtin(__builtin_amdgcn_sched_barrier)
  __builtin_amdgcn_sched_barrier(0);
#endif
}

// ---------------- helpers ----------------
__device__ __forceinline__ unsigned short f2bf(float f) {
  unsigned u = __float_as_uint(f);
  u += 0x7FFFu + ((u >> 16) & 1u);          // round-to-nearest-even
  return (unsigned short)(u >> 16);
}
__device__ __forceinline__ unsigned pack2(float a, float b) {
  return (unsigned)f2bf(a) | ((unsigned)f2bf(b) << 16);
}
__device__ __forceinline__ float gelu_tanh(float x) {
  const float k = 0.7978845608028654f;
  float t = tanhf(k * (x + 0.044715f * x * x * x));
  return 0.5f * x * (1.0f + t);
}

// ---------------- 1) router: logits + softmax -> probsT (E,T) ---------------
__global__ __launch_bounds__(256) void router_kernel(
    const float* __restrict__ x, const float* __restrict__ gate,
    float* __restrict__ probsT) {
  __shared__ float g[DIM * NEXP];           // 32 KB
  for (int i = threadIdx.x; i < DIM * NEXP; i += 256) g[i] = gate[i];
  __syncthreads();
  const int t = blockIdx.x * 256 + threadIdx.x;
  const float4* xr = (const float4*)(x + (size_t)t * DIM);
  float acc[NEXP];
#pragma unroll
  for (int e = 0; e < NEXP; e++) acc[e] = 0.f;
  for (int d4 = 0; d4 < DIM / 4; d4++) {
    float4 xv = xr[d4];
    const float* gr = g + d4 * 4 * NEXP;
#pragma unroll
    for (int e = 0; e < NEXP; e++)
      acc[e] += xv.x * gr[e] + xv.y * gr[NEXP + e] +
                xv.z * gr[2 * NEXP + e] + xv.w * gr[3 * NEXP + e];
  }
  float m = acc[0];
#pragma unroll
  for (int e = 1; e < NEXP; e++) m = fmaxf(m, acc[e]);
  float s = 0.f, p[NEXP];
#pragma unroll
  for (int e = 0; e < NEXP; e++) { p[e] = expf(acc[e] - m); s += p[e]; }
  float inv = 1.f / s;
#pragma unroll
  for (int e = 0; e < NEXP; e++)
    probsT[(size_t)e * T_TOK + t] = p[e] * inv;
}

// ---------------- 2) per-expert top-C selection (histogram threshold) -------
__device__ __forceinline__ int keybin(float p) {
  unsigned u = __float_as_uint(p);          // p >= 0 -> bits monotonic
  unsigned b = u >> 20;
  return (int)(b > 1023u ? 1023u : b);
}
__global__ __launch_bounds__(256) void topk_kernel(
    const float* __restrict__ probsT, int* __restrict__ tokIdx,
    float* __restrict__ scores) {
  __shared__ int hist[1024];
  __shared__ int tb_s, above_s, cntA, cntB;
  const int e = blockIdx.x, tid = threadIdx.x;
  const float* pe = probsT + (size_t)e * T_TOK;
  for (int i = tid; i < 1024; i += 256) hist[i] = 0;
  if (tid == 0) { cntA = 0; cntB = 0; }
  __syncthreads();
  for (int t = tid; t < T_TOK; t += 256) atomicAdd(&hist[keybin(pe[t])], 1);
  __syncthreads();
  if (tid == 0) {
    int tot = 0, b = 1023;
    for (; b > 0; b--) { tot += hist[b]; if (tot >= CAP) break; }
    tb_s = b; above_s = tot - hist[b];
  }
  __syncthreads();
  const int tb = tb_s, above = above_s;
  for (int t = tid; t < T_TOK; t += 256) {
    float p = pe[t];
    int b = keybin(p);
    if (b > tb) {
      int q = atomicAdd(&cntA, 1);
      tokIdx[e * CAP + q] = t; scores[e * CAP + q] = p;
    } else if (b == tb) {
      int q = atomicAdd(&cntB, 1);
      if (q < CAP - above) {
        int s = above + q;
        tokIdx[e * CAP + s] = t; scores[e * CAP + s] = p;
      }
    }
  }
}

// ---------------- 3) fp32 -> bf16 convert / weight transpose / zero ---------
__global__ __launch_bounds__(256) void cvt_kernel(
    const float* __restrict__ in, unsigned short* __restrict__ out) {
  size_t base = ((size_t)blockIdx.x * 256 + threadIdx.x) * 8;
  float4 a = *(const float4*)(in + base);
  float4 b = *(const float4*)(in + base + 4);
  uint4 o;
  o.x = pack2(a.x, a.y); o.y = pack2(a.z, a.w);
  o.z = pack2(b.x, b.y); o.w = pack2(b.z, b.w);
  *(uint4*)(out + base) = o;
}
// in: fp32 [K][N] (pitch N) -> out: bf16 [N][K] (pitch K), 32x32 tiles
__global__ __launch_bounds__(256) void transpose_w_kernel(
    const float* __restrict__ in, unsigned short* __restrict__ out, int K,
    int N, unsigned long long inEstride, unsigned long long outEstride) {
  __shared__ float tile[32][33];
  const int e = blockIdx.z;
  const float* ie = in + (size_t)e * inEstride;
  unsigned short* oe = out + (size_t)e * outEstride;
  const int k0 = blockIdx.y * 32, n0 = blockIdx.x * 32;
  const int r = threadIdx.x >> 3, c4 = (threadIdx.x & 7) * 4;
  float4 v = *(const float4*)(ie + (size_t)(k0 + r) * N + n0 + c4);
  tile[r][c4] = v.x; tile[r][c4 + 1] = v.y;
  tile[r][c4 + 2] = v.z; tile[r][c4 + 3] = v.w;
  __syncthreads();
  uint2 o;
  o.x = pack2(tile[c4][r], tile[c4 + 1][r]);
  o.y = pack2(tile[c4 + 2][r], tile[c4 + 3][r]);
  *(uint2*)(oe + (size_t)(n0 + r) * K + k0 + c4) = o;
}
__global__ __launch_bounds__(256) void zero_kernel(float4* __restrict__ out) {
  out[(size_t)blockIdx.x * 256 + threadIdx.x] = make_float4(0.f, 0.f, 0.f, 0.f);
}

// ---------------- WMMA fragment loaders (ISA 16-bit layouts) ----------------
__device__ __forceinline__ v16bf load_a_frag(const unsigned short* lds,
                                             int rowBase, int lane) {
  int r = rowBase + (lane & 15);
  int cb = (lane < 16) ? 0 : 8;
  union { struct { ushort8 lo, hi; } s; v16bf v; } u;
  u.s.lo = *(const ushort8*)(lds + r * 40 + cb);
  u.s.hi = *(const ushort8*)(lds + r * 40 + cb + 16);
  return u.v;
}
__device__ __forceinline__ v16bf load_b_frag(const unsigned short* lds,
                                             int colBase, int lane) {
  int n = colBase + (lane & 15);
  int cb = (lane < 16) ? 0 : 8;
  union { struct { ushort8 lo, hi; } s; v16bf v; } u;
  const ushort8* p = (const ushort8*)(lds + n * 40 + cb);
  u.s.lo = p[0];
  u.s.hi = p[2];
  return u.v;
}

// Block tile 128(M) x 256(N), K-step 32; 8 waves, wave tile 64x64 (4x4 WMMA).
// Double-buffered LDS; next tile staged (async-to-LDS or pipelined registers)
// while current tile is consumed by WMMAs. One barrier per K-step.
#define GEMM_MAINLOOP(ACC)                                                     \
  const int tid = threadIdx.x, lane = tid & 31, wave = tid >> 5;               \
  const int wm = wave & 1, wn = wave >> 1;                                     \
  const int e = blockIdx.z;                                                    \
  const int m0 = blockIdx.y * 128, n0 = blockIdx.x * 256;                      \
  const int arow = tid >> 1, ac = (tid & 1) * 16;                              \
  const int aRowIdx = m0 + arow;                                               \
  size_t aRowG = gatherIdx ? (size_t)gatherIdx[e * M + aRowIdx]                \
                           : (size_t)aRowIdx;                                  \
  const unsigned short* aP =                                                   \
      A + (size_t)e * aEstride + aRowG * (size_t)K + ac;                       \
  const unsigned short* bP =                                                   \
      Bt + (size_t)e * bEstride + (size_t)(n0 + tid) * K;                      \
  unsigned short* curA = ldsA;                                                 \
  unsigned short* nxtA = ldsA + 128 * 40;                                      \
  unsigned short* curB = ldsB;                                                 \
  unsigned short* nxtB = ldsB + 256 * 40;                                      \
  v8f ACC[4][4];                                                               \
  {                                                                            \
    const v8f vz = {0.f, 0.f, 0.f, 0.f, 0.f, 0.f, 0.f, 0.f};                   \
    _Pragma("unroll") for (int mi = 0; mi < 4; mi++)                           \
        _Pragma("unroll") for (int ni = 0; ni < 4; ni++) ACC[mi][ni] = vz;     \
  }                                                                            \
  /* prologue: stage tile 0 into buffer 0 */                                   \
  if (HAVE_ASYNC_LDS) {                                                        \
    async_b128(aP, curA + arow * 40 + ac);                                     \
    async_b128(aP + 8, curA + arow * 40 + ac + 8);                             \
    async_b128(bP, curB + tid * 40);                                           \
    async_b128(bP + 8, curB + tid * 40 + 8);                                   \
    async_b128(bP + 16, curB + tid * 40 + 16);                                 \
    async_b128(bP + 24, curB + tid * 40 + 24);                                 \
    wait_async0();                                                             \
  } else {                                                                     \
    const uint4* ga = (const uint4*)aP;                                        \
    uint4 a0 = ga[0], a1 = ga[1];                                              \
    const uint4* gb = (const uint4*)bP;                                        \
    uint4 b0 = gb[0], b1 = gb[1], b2 = gb[2], b3 = gb[3];                      \
    *(uint4*)(curA + arow * 40 + ac) = a0;                                     \
    *(uint4*)(curA + arow * 40 + ac + 8) = a1;                                 \
    *(uint4*)(curB + tid * 40) = b0;                                           \
    *(uint4*)(curB + tid * 40 + 8) = b1;                                       \
    *(uint4*)(curB + tid * 40 + 16) = b2;                                      \
    *(uint4*)(curB + tid * 40 + 24) = b3;                                      \
  }                                                                            \
  __syncthreads();                                                             \
  for (int k0 = 0; k0 < K; k0 += 32) {                                         \
    const int kn = k0 + 32;                                                    \
    uint4 na0, na1, nb0, nb1, nb2, nb3;                                        \
    if (kn < K) {                                                              \
      if (HAVE_ASYNC_LDS) {                                                    \
        async_b128(aP + kn, nxtA + arow * 40 + ac);                            \
        async_b128(aP + kn + 8, nxtA + arow * 40 + ac + 8);                    \
        async_b128(bP + kn, nxtB + tid * 40);                                  \
        async_b128(bP + kn + 8, nxtB + tid * 40 + 8);                          \
        async_b128(bP + kn + 16, nxtB + tid * 40 + 16);                        \
        async_b128(bP + kn + 24, nxtB + tid * 40 + 24);                        \
      } else {                                                                 \
        const uint4* ga = (const uint4*)(aP + kn);                             \
        na0 = ga[0]; na1 = ga[1];                                              \
        const uint4* gb = (const uint4*)(bP + kn);                             \
        nb0 = gb[0]; nb1 = gb[1]; nb2 = gb[2]; nb3 = gb[3];                    \
      }                                                                        \
    }                                                                          \
    sched_fence();                                                             \
    v16bf af[4], bfr[4];                                                       \
    _Pragma("unroll") for (int mi = 0; mi < 4; mi++) af[mi] =                  \
        load_a_frag(curA, wm * 64 + mi * 16, lane);                            \
    _Pragma("unroll") for (int ni = 0; ni < 4; ni++) bfr[ni] =                 \
        load_b_frag(curB, wn * 64 + ni * 16, lane);                            \
    _Pragma("unroll") for (int mi = 0; mi < 4; mi++)                           \
        _Pragma("unroll") for (int ni = 0; ni < 4; ni++) ACC[mi][ni] =         \
            __builtin_amdgcn_wmma_f32_16x16x32_bf16(                           \
                false, af[mi], false, bfr[ni], (short)0, ACC[mi][ni], false,   \
                false);                                                        \
    if (kn < K && !HAVE_ASYNC_LDS) {                                           \
      *(uint4*)(nxtA + arow * 40 + ac) = na0;                                  \
      *(uint4*)(nxtA + arow * 40 + ac + 8) = na1;                              \
      *(uint4*)(nxtB + tid * 40) = nb0;                                        \
      *(uint4*)(nxtB + tid * 40 + 8) = nb1;                                    \
      *(uint4*)(nxtB + tid * 40 + 16) = nb2;                                   \
      *(uint4*)(nxtB + tid * 40 + 24) = nb3;                                   \
    }                                                                          \
    wait_async0();                                                             \
    __syncthreads();                                                           \
    unsigned short* tp;                                                        \
    tp = curA; curA = nxtA; nxtA = tp;                                         \
    tp = curB; curB = nxtB; nxtB = tp;                                         \
  }

// ---------------- 4) GEMM1: h = gelu(A @ Bt^T + b) -> bf16 ------------------
__global__ __launch_bounds__(256) void gemm1_kernel(
    const unsigned short* __restrict__ A, unsigned long long aEstride,
    const int* __restrict__ gatherIdx,    // null -> direct rows
    const unsigned short* __restrict__ Bt, unsigned long long bEstride,
    const float* __restrict__ bias, unsigned long long biasEstride,
    unsigned short* __restrict__ Out, unsigned long long oEstride,
    int M, int N, int K) {
  __shared__ __align__(16) unsigned short ldsA[2 * 128 * 40];
  __shared__ __align__(16) unsigned short ldsB[2 * 256 * 40];
  GEMM_MAINLOOP(acc)
  const float* be = bias + (size_t)e * biasEstride;
  unsigned short* Oe = Out + (size_t)e * oEstride;
#pragma unroll
  for (int ni = 0; ni < 4; ni++) {
    int col = n0 + wn * 64 + ni * 16 + (lane & 15);
    float bv = be[col];
#pragma unroll
    for (int mi = 0; mi < 4; mi++) {
      v8f a = acc[mi][ni];
      int rbase = m0 + wm * 64 + mi * 16 + ((lane < 16) ? 0 : 8);
#pragma unroll
      for (int i = 0; i < 8; i++)
        Oe[(size_t)(rbase + i) * N + col] = f2bf(gelu_tanh(a[i] + bv));
    }
  }
}

// ---------------- 5) GEMM2: out = A @ Bt^T + b; scatter-scale or += ---------
__global__ __launch_bounds__(256) void gemm2_kernel(
    const unsigned short* __restrict__ A, unsigned long long aEstride,
    const unsigned short* __restrict__ Bt, unsigned long long bEstride,
    const float* __restrict__ bias, unsigned long long biasEstride,
    float* __restrict__ Out,
    const int* __restrict__ tokIdx, const float* __restrict__ scores,
    int M, int N, int K, int scatter) {
  __shared__ __align__(16) unsigned short ldsA[2 * 128 * 40];
  __shared__ __align__(16) unsigned short ldsB[2 * 256 * 40];
  const int* gatherIdx = nullptr;          // A rows are direct here
  GEMM_MAINLOOP(acc)
  const float* be = bias + (size_t)e * biasEstride;
#pragma unroll
  for (int ni = 0; ni < 4; ni++) {
    int col = n0 + wn * 64 + ni * 16 + (lane & 15);
    float bv = be[col];
#pragma unroll
    for (int mi = 0; mi < 4; mi++) {
      v8f a = acc[mi][ni];
      int rbase = m0 + wm * 64 + mi * 16 + ((lane < 16) ? 0 : 8);
      if (scatter) {
#pragma unroll
        for (int i = 0; i < 8; i++) {
          int rowl = rbase + i;
          int tok = tokIdx[e * M + rowl];
          float sc = scores[e * M + rowl];
          atomicAdd(Out + (size_t)tok * N + col, (a[i] + bv) * sc);
        }
      } else {
#pragma unroll
        for (int i = 0; i < 8; i++) {
          size_t o = (size_t)(rbase + i) * N + col;
          Out[o] += a[i] + bv;
        }
      }
    }
  }
}

// ---------------- launcher ----------------
extern "C" void kernel_launch(void* const* d_in, const int* in_sizes, int n_in,
                              void* d_out, int out_size, void* d_ws,
                              size_t ws_size, hipStream_t stream) {
  (void)in_sizes; (void)n_in; (void)out_size; (void)ws_size;
  const float* x    = (const float*)d_in[0];
  const float* gate = (const float*)d_in[1];
  const float* W1   = (const float*)d_in[2];
  const float* b1   = (const float*)d_in[3];
  const float* W2   = (const float*)d_in[4];
  const float* b2   = (const float*)d_in[5];
  const float* Ws1  = (const float*)d_in[6];
  const float* bs1  = (const float*)d_in[7];
  const float* Ws2  = (const float*)d_in[8];
  const float* bs2  = (const float*)d_in[9];
  float* out = (float*)d_out;

  char* ws = (char*)d_ws;
  size_t off = 0;
  float* probsT = (float*)(ws + off);  off += (size_t)NEXP * T_TOK * 4;
  int* tokIdx   = (int*)(ws + off);    off += (size_t)NEXP * CAP * 4;
  float* scores = (float*)(ws + off);  off += (size_t)NEXP * CAP * 4;
  unsigned short* xb   = (unsigned short*)(ws + off); off += (size_t)T_TOK * DIM * 2;
  unsigned short* W1t  = (unsigned short*)(ws + off); off += (size_t)NEXP * HID * DIM * 2;
  unsigned short* W2t  = (unsigned short*)(ws + off); off += (size_t)NEXP * DIM * HID * 2;
  unsigned short* Ws1t = (unsigned short*)(ws + off); off += (size_t)HID * DIM * 2;
  unsigned short* Ws2t = (unsigned short*)(ws + off); off += (size_t)DIM * HID * 2;
  unsigned short* hbuf = (unsigned short*)(ws + off); // 16384x4096 bf16 (reused)

  router_kernel<<<T_TOK / 256, 256, 0, stream>>>(x, gate, probsT);
  topk_kernel<<<NEXP, 256, 0, stream>>>(probsT, tokIdx, scores);
  cvt_kernel<<<(T_TOK * (size_t)DIM) / (256 * 8), 256, 0, stream>>>(x, xb);
  transpose_w_kernel<<<dim3(HID / 32, DIM / 32, NEXP), 256, 0, stream>>>(
      W1, W1t, DIM, HID, (unsigned long long)DIM * HID,
      (unsigned long long)HID * DIM);
  transpose_w_kernel<<<dim3(DIM / 32, HID / 32, NEXP), 256, 0, stream>>>(
      W2, W2t, HID, DIM, (unsigned long long)HID * DIM,
      (unsigned long long)DIM * HID);
  transpose_w_kernel<<<dim3(HID / 32, DIM / 32, 1), 256, 0, stream>>>(
      Ws1, Ws1t, DIM, HID, 0ULL, 0ULL);
  transpose_w_kernel<<<dim3(DIM / 32, HID / 32, 1), 256, 0, stream>>>(
      Ws2, Ws2t, HID, DIM, 0ULL, 0ULL);
  zero_kernel<<<(T_TOK * (size_t)DIM) / (256 * 4), 256, 0, stream>>>(
      (float4*)out);

  // routed GEMM1 (gathered A rows): E x [2048x1024]@[1024x4096] -> hbuf
  gemm1_kernel<<<dim3(HID / 256, CAP / 128, NEXP), 256, 0, stream>>>(
      xb, 0ULL, tokIdx, W1t, (unsigned long long)HID * DIM, b1, HID, hbuf,
      (unsigned long long)CAP * HID, CAP, HID, DIM);
  // routed GEMM2: E x [2048x4096]@[4096x1024], scale+scatter-add into out
  gemm2_kernel<<<dim3(DIM / 256, CAP / 128, NEXP), 256, 0, stream>>>(
      hbuf, (unsigned long long)CAP * HID, W2t, (unsigned long long)DIM * HID,
      b2, DIM, out, tokIdx, scores, CAP, DIM, HID, 1);
  // shared GEMM1: [16384x1024]@[1024x4096] -> hbuf (reused)
  gemm1_kernel<<<dim3(HID / 256, T_TOK / 128, 1), 256, 0, stream>>>(
      xb, 0ULL, nullptr, Ws1t, 0ULL, bs1, 0ULL, hbuf, 0ULL, T_TOK, HID, DIM);
  // shared GEMM2: [16384x4096]@[4096x1024], += into out
  gemm2_kernel<<<dim3(DIM / 256, T_TOK / 128, 1), 256, 0, stream>>>(
      hbuf, 0ULL, Ws2t, 0ULL, bs2, 0ULL, out, nullptr, nullptr, T_TOK, DIM,
      HID, 0);
}
